// ViT_32332513804420
// MI455X (gfx1250) — matrix operand order repaired and checked by reading, not compiled, not run on hardware
//
#include <hip/hip_runtime.h>
#include <hip/hip_bf16.h>
#include <math.h>

// ---------------------------------------------------------------------------
// ViT forward for MI455X (gfx1250).  All GEMMs run through one generic
// wave32 WMMA kernel: bf16 operands staged in LDS, software-pipelined global
// prefetch, f32 accumulation with v_wmma_f32_16x16x32_bf16, fused
// LayerNorm-on-A / bias / residual / GELU epilogues.
// Block tile 128x128 (BK=32), 8 waves in a 4(M) x 2(N) grid, each wave a
// 32x64 tile = 2x4 WMMA tiles -> 8 WMMAs per 12 ds_load_b128 per K-step.
// ---------------------------------------------------------------------------

typedef __attribute__((ext_vector_type(16))) __bf16 v16bf;
typedef __attribute__((ext_vector_type(8)))  float  v8f;

union FragBF { uint4 u[2]; v16bf v; };

__device__ __forceinline__ unsigned int bf16bits(float f) {
  unsigned int u = __float_as_uint(f);
  return (u + 0x7fffu + ((u >> 16) & 1u)) >> 16;   // round-to-nearest-even
}
__device__ __forceinline__ unsigned int pack2bf(float lo, float hi) {
  return bf16bits(lo) | (bf16bits(hi) << 16);
}

__device__ __forceinline__ float gelu_exact(float x) {
  return 0.5f * x * (1.0f + erff(x * 0.70710678118654752f));
}

// ---------------------------------------------------------------------------
// Generic batched GEMM: C[z] = op(A[z]) * B[z] + bias (+residual) (+GELU)
//   A: [M,K] f32 (optionally LayerNorm'd per-row on load, mu/rs indexed by row)
//   B: [K,N] f32 (or [N,K] when TRB)
// Per-batch offsets (two-term):  X += (z / xDiv) * xStr1 + (z % xMod) * xStr2
// ---------------------------------------------------------------------------
template<bool TRB, bool LNA, bool DG, bool DR>
__global__ __launch_bounds__(256) void vit_gemm_wmma(
    const float* __restrict__ A, long long lda, long long aDiv, long long aStr1,
    long long aMod, long long aStr2,
    const float* __restrict__ Bm, long long ldb, long long bDiv, long long bStr1,
    long long bMod, long long bStr2,
    float* __restrict__ C, long long ldc, long long cDiv, long long cStr1,
    long long cMod, long long cStr2,
    const float* __restrict__ bias, long long biasMod, long long biasStr,
    const float* __restrict__ resid,
    const float* __restrict__ mu, const float* __restrict__ rs,
    const float* __restrict__ lnG, const float* __restrict__ lnB, long long lnMod,
    int Mdim, int Ndim, int Kdim)
{
  // 80-byte padded rows: 16B aligned for ds_load_b128, no bank conflicts.
  __shared__ __align__(16) unsigned short As[128 * 40];
  __shared__ __align__(16) unsigned short Bs[128 * 40];

  const int tid = threadIdx.x;
  const int z  = blockIdx.z;
  const int m0 = blockIdx.y * 128;
  const int n0 = blockIdx.x * 128;

  const float* Ap = A + (long long)(z / aDiv) * aStr1 + (long long)(z % aMod) * aStr2;
  const float* Bp = Bm + (long long)(z / bDiv) * bStr1 + (long long)(z % bMod) * bStr2;
  const long long coff = (long long)(z / cDiv) * cStr1 + (long long)(z % cMod) * cStr2;
  float* Cp = C + coff;
  const float* Rp = DR ? (resid + coff) : nullptr;

  long long lnoff = 0, boff = 0;
  if (LNA) lnoff = (long long)(z % lnMod) * (long long)Kdim;
  if (bias) boff = (long long)(z % biasMod) * biasStr;

  const int wave = tid >> 5;
  const int lane = tid & 31;
  const int wy = wave >> 1;      // 0..3  (M, 32 rows each)
  const int wx = wave & 1;       // 0..1  (N, 64 cols each)
  const int l16 = lane & 15;
  const int hi  = lane >> 4;

  // per-thread staging coordinates (16 K-elements per thread for both tiles)
  const int arow = tid >> 1;               // 0..127
  const int akc  = (tid & 1) << 4;         // 0,16
  const int btn  = TRB ? (tid >> 1) : (tid & 127);              // 0..127
  const int btkc = TRB ? ((tid & 1) << 4) : ((tid >> 7) << 4);  // 0,16

  const bool fullM = (m0 + 128) <= Mdim;
  const bool fullN = (n0 + 128) <= Ndim;

  // ------ staging helpers ------
  auto loadA_fast = [&](int kt, float (&v)[16]) {
    int grow = m0 + arow;
    const float* ap = Ap + (long long)grow * lda + (kt + akc);
    #pragma unroll
    for (int j = 0; j < 4; ++j) {
      float4 t = *reinterpret_cast<const float4*>(ap + 4 * j);
      v[4*j+0]=t.x; v[4*j+1]=t.y; v[4*j+2]=t.z; v[4*j+3]=t.w;
    }
    if (LNA) {
      float mm = mu[grow], rr = rs[grow];
      #pragma unroll
      for (int i = 0; i < 16; ++i) {
        int kk = kt + akc + i;
        v[i] = (v[i] - mm) * rr * lnG[lnoff + kk] + lnB[lnoff + kk];
      }
    }
  };
  auto storeA = [&](float (&v)[16]) {
    uint4 w0, w1;
    w0.x = pack2bf(v[0],  v[1]);
    w0.y = pack2bf(v[2],  v[3]);
    w0.z = pack2bf(v[4],  v[5]);
    w0.w = pack2bf(v[6],  v[7]);
    w1.x = pack2bf(v[8],  v[9]);
    w1.y = pack2bf(v[10], v[11]);
    w1.z = pack2bf(v[12], v[13]);
    w1.w = pack2bf(v[14], v[15]);
    *reinterpret_cast<uint4*>(&As[arow * 40 + akc]) = w0;
    *reinterpret_cast<uint4*>(&As[arow * 40 + akc + 8]) = w1;
  };
  auto slowA = [&](int kt) {   // edge tiles: rolled, clamped, direct b16 stores
    int grow = m0 + arow;
    int growc = grow < Mdim ? grow : (Mdim - 1);
    const float* ap = Ap + (long long)growc * lda;
    float mm = 0.f, rr = 1.f;
    if (LNA) { mm = mu[growc]; rr = rs[growc]; }
    #pragma clang loop unroll(disable)
    for (int i = 0; i < 16; ++i) {
      int kk = kt + akc + i;
      int kkc = kk < Kdim ? kk : (Kdim - 1);
      float v = ap[kkc];
      if (LNA) v = (v - mm) * rr * lnG[lnoff + kkc] + lnB[lnoff + kkc];
      v = (grow < Mdim && kk < Kdim) ? v : 0.f;
      As[arow * 40 + akc + i] = (unsigned short)bf16bits(v);
    }
  };
  auto loadB_fast = [&](int kt, float (&v)[16]) {
    int gn = n0 + btn;
    if (TRB) {
      const float* bp = Bp + (long long)gn * ldb + (kt + btkc);
      #pragma unroll
      for (int j = 0; j < 4; ++j) {
        float4 t = *reinterpret_cast<const float4*>(bp + 4 * j);
        v[4*j+0]=t.x; v[4*j+1]=t.y; v[4*j+2]=t.z; v[4*j+3]=t.w;
      }
    } else {
      #pragma unroll
      for (int i = 0; i < 16; ++i)
        v[i] = Bp[(long long)(kt + btkc + i) * ldb + gn];
    }
  };
  auto storeB = [&](float (&v)[16]) {
    uint4 w0, w1;
    w0.x = pack2bf(v[0],  v[1]);
    w0.y = pack2bf(v[2],  v[3]);
    w0.z = pack2bf(v[4],  v[5]);
    w0.w = pack2bf(v[6],  v[7]);
    w1.x = pack2bf(v[8],  v[9]);
    w1.y = pack2bf(v[10], v[11]);
    w1.z = pack2bf(v[12], v[13]);
    w1.w = pack2bf(v[14], v[15]);
    *reinterpret_cast<uint4*>(&Bs[btn * 40 + btkc]) = w0;
    *reinterpret_cast<uint4*>(&Bs[btn * 40 + btkc + 8]) = w1;
  };
  auto slowB = [&](int kt) {
    int gn = n0 + btn;
    int gnc = gn < Ndim ? gn : (Ndim - 1);
    #pragma clang loop unroll(disable)
    for (int i = 0; i < 16; ++i) {
      int kk = kt + btkc + i;
      int kkc = kk < Kdim ? kk : (Kdim - 1);
      float v = TRB ? Bp[(long long)gnc * ldb + kkc]
                    : Bp[(long long)kkc * ldb + gnc];
      v = (gn < Ndim && kk < Kdim) ? v : 0.f;
      Bs[btn * 40 + btkc + i] = (unsigned short)bf16bits(v);
    }
  };

  v8f acc[2][4];
  #pragma unroll
  for (int mi = 0; mi < 2; ++mi)
    #pragma unroll
    for (int ni = 0; ni < 4; ++ni)
      acc[mi][ni] = (v8f){0.f,0.f,0.f,0.f,0.f,0.f,0.f,0.f};

  float aV[16];
  float bV[16];
  if (fullM && 32 <= Kdim) loadA_fast(0, aV);
  if (fullN && 32 <= Kdim) loadB_fast(0, bV);

  for (int kt = 0; kt < Kdim; kt += 32) {
    const bool aFull = fullM && (kt + 32) <= Kdim;
    const bool bFull = fullN && (kt + 32) <= Kdim;
    __syncthreads();
    if (aFull) storeA(aV); else slowA(kt);
    if (bFull) storeB(bV); else slowB(kt);
    __syncthreads();

    // prefetch next K-tile into registers; overlaps with the WMMAs below
    int ktn = kt + 32;
    if (ktn < Kdim) {
      if (fullM && (ktn + 32) <= Kdim) loadA_fast(ktn, aV);
      if (fullN && (ktn + 32) <= Kdim) loadB_fast(ktn, bV);
    }

    // ---- fragments per the CDNA5 16-bit WMMA layouts ----
    // A 16x32: lanes 0-15 K {0-7,16-23}, lanes 16-31 K {8-15,24-31}
    FragBF afr[2], bfr[4];
    #pragma unroll
    for (int mi = 0; mi < 2; ++mi) {
      int row = wy * 32 + mi * 16 + l16;
      afr[mi].u[0] = *reinterpret_cast<const uint4*>(&As[row * 40 + hi * 8]);
      afr[mi].u[1] = *reinterpret_cast<const uint4*>(&As[row * 40 + hi * 8 + 16]);
    }
    // B 32x16: lanes 0-15 K {0-7,16-23}, lanes 16-31 K {8-15,24-31}
    #pragma unroll
    for (int ni = 0; ni < 4; ++ni) {
      int rn = wx * 64 + ni * 16 + l16;
      const uint4* p = reinterpret_cast<const uint4*>(&Bs[rn * 40 + hi * 8]);
      bfr[ni].u[0] = p[0];
      bfr[ni].u[1] = p[2];
    }
    #pragma unroll
    for (int mi = 0; mi < 2; ++mi)
      #pragma unroll
      for (int ni = 0; ni < 4; ++ni)
        acc[mi][ni] = __builtin_amdgcn_wmma_f32_16x16x32_bf16(
            false, afr[mi].v, false, bfr[ni].v, (short)0, acc[mi][ni],
            false, false);
  }

  // ---- epilogue: C layout (vgpr i, lane) -> m = i + 8*(lane/16), n = lane%16
  if (fullM && fullN) {
    #pragma unroll
    for (int mi = 0; mi < 2; ++mi) {
      #pragma unroll
      for (int ni = 0; ni < 4; ++ni) {
        int col = n0 + wx * 64 + ni * 16 + l16;
        float bvv = bias ? bias[boff + col] : 0.f;
        #pragma unroll
        for (int i = 0; i < 8; ++i) {
          int row = m0 + wy * 32 + mi * 16 + hi * 8 + i;
          float v = acc[mi][ni][i] + bvv;
          if (DR) v += Rp[(long long)row * ldc + col];
          if (DG) v = gelu_exact(v);
          Cp[(long long)row * ldc + col] = v;
        }
      }
    }
  } else {
    #pragma unroll
    for (int mi = 0; mi < 2; ++mi) {
      #pragma unroll
      for (int ni = 0; ni < 4; ++ni) {
        int col = n0 + wx * 64 + ni * 16 + l16;
        if (col >= Ndim) continue;
        float bvv = bias ? bias[boff + col] : 0.f;
        #pragma unroll
        for (int i = 0; i < 8; ++i) {
          int row = m0 + wy * 32 + mi * 16 + hi * 8 + i;
          if (row >= Mdim) continue;
          float v = acc[mi][ni][i] + bvv;
          if (DR) v += Rp[(long long)row * ldc + col];
          if (DG) v = gelu_exact(v);
          Cp[(long long)row * ldc + col] = v;
        }
      }
    }
  }
}

// ---------------------------------------------------------------------------
// im2col for stride-16 patch conv: rows = (b, n=h*14+w), cols = c*256+p*16+q
// ---------------------------------------------------------------------------
__global__ void vit_im2col(const float* __restrict__ x, float* __restrict__ Ape) {
  long long idx = (long long)blockIdx.x * blockDim.x + threadIdx.x;
  if (idx >= 3136LL * 768) return;
  int col = (int)(idx % 768);
  long long r = idx / 768;
  int b = (int)(r / 196);
  int n = (int)(r % 196);
  int c   = col >> 8;
  int rem = col & 255;
  int p = rem >> 4, q = rem & 15;
  int ri = (n / 14) * 16 + p;
  int ci = (n % 14) * 16 + q;
  Ape[idx] = x[(((long long)b * 3 + c) * 224 + ri) * 224 + ci];
}

// h[b,s,:] = (s==0 ? cls : patch_out[b,s-1,:]) + pos[s,:]
__global__ void vit_assemble(const float* __restrict__ pe, const float* __restrict__ cls,
                             const float* __restrict__ pos, float* __restrict__ h) {
  long long idx = (long long)blockIdx.x * blockDim.x + threadIdx.x;
  if (idx >= 3152LL * 768) return;
  int d = (int)(idx % 768);
  long long r = idx / 768;
  int b = (int)(r / 197);
  int s = (int)(r % 197);
  float v = (s == 0) ? cls[d] : pe[((long long)b * 196 + (s - 1)) * 768 + d];
  h[idx] = v + pos[(long long)s * 768 + d];
}

// per-token mean / rsqrt(var+eps) over D=768; rowStride in rows (197 for CLS)
__global__ __launch_bounds__(256) void vit_ln_stats(const float* __restrict__ X,
    long long rowStride, float* __restrict__ mu, float* __restrict__ rs) {
  __shared__ float red[256];
  int r = blockIdx.x;
  const float* x = X + (long long)r * rowStride * 768;
  float s = 0.f, s2 = 0.f;
  for (int i = threadIdx.x; i < 768; i += 256) { float v = x[i]; s += v; s2 += v * v; }
  red[threadIdx.x] = s; __syncthreads();
  for (int off = 128; off > 0; off >>= 1) {
    if (threadIdx.x < off) red[threadIdx.x] += red[threadIdx.x + off];
    __syncthreads();
  }
  float sum = red[0]; __syncthreads();
  red[threadIdx.x] = s2; __syncthreads();
  for (int off = 128; off > 0; off >>= 1) {
    if (threadIdx.x < off) red[threadIdx.x] += red[threadIdx.x + off];
    __syncthreads();
  }
  if (threadIdx.x == 0) {
    float m = sum * (1.f / 768.f);
    float var = red[0] * (1.f / 768.f) - m * m;
    mu[r] = m;
    rs[r] = rsqrtf(var + 1e-5f);
  }
}

// softmax over the QUERY axis (reference quirk): per (z, k) column of scores
__global__ __launch_bounds__(256) void vit_softmax_q(
    const float* __restrict__ Sc, float* __restrict__ Pr,
    int Sdim, int ld, float scale) {
  int z = blockIdx.y;
  int k = blockIdx.x * blockDim.x + threadIdx.x;
  if (k >= Sdim) return;
  const float* sp = Sc + (long long)z * Sdim * ld + k;
  float*       pp = Pr + (long long)z * Sdim * ld + k;
  float mx = -3.0e38f;
  for (int qi = 0; qi < Sdim; ++qi) mx = fmaxf(mx, sp[(long long)qi * ld] * scale);
  float sum = 0.f;
  for (int qi = 0; qi < Sdim; ++qi) sum += __expf(sp[(long long)qi * ld] * scale - mx);
  float inv = 1.f / sum;
  for (int qi = 0; qi < Sdim; ++qi)
    pp[(long long)qi * ld] = __expf(sp[(long long)qi * ld] * scale - mx) * inv;
}

// final LN on CLS token + tiny 768x4 classifier
__global__ void vit_classifier(const float* __restrict__ h, const float* __restrict__ mu,
    const float* __restrict__ rs, const float* __restrict__ g, const float* __restrict__ b,
    const float* __restrict__ Wc, const float* __restrict__ bc, float* __restrict__ out) {
  int t = threadIdx.x;
  if (t >= 64) return;
  int bi = t >> 2, c = t & 3;
  const float* x = h + (long long)bi * 197 * 768;
  float m = mu[bi], r = rs[bi];
  float acc = 0.f;
  for (int d = 0; d < 768; ++d) {
    float xn = (x[d] - m) * r * g[d] + b[d];
    acc += xn * Wc[d * 4 + c];
  }
  out[t] = acc + bc[c];
}

// ---------------------------------------------------------------------------
template<bool TRB, bool LNA, bool DG, bool DR>
static inline void launch_gemm(hipStream_t st,
    const float* A, long long lda, long long aDiv, long long aStr1,
    long long aMod, long long aStr2,
    const float* Bm, long long ldb, long long bDiv, long long bStr1,
    long long bMod, long long bStr2,
    float* C, long long ldc, long long cDiv, long long cStr1,
    long long cMod, long long cStr2,
    const float* bias, long long biasMod, long long biasStr,
    const float* res,
    const float* mu, const float* rs,
    const float* lnG, const float* lnB, long long lnMod,
    int M, int N, int K, int Z)
{
  dim3 grid((unsigned)((N + 127) / 128), (unsigned)((M + 127) / 128), (unsigned)Z);
  vit_gemm_wmma<TRB, LNA, DG, DR><<<grid, 256, 0, st>>>(
      A, lda, aDiv, aStr1, aMod, aStr2,
      Bm, ldb, bDiv, bStr1, bMod, bStr2,
      C, ldc, cDiv, cStr1, cMod, cStr2,
      bias, biasMod, biasStr, res, mu, rs, lnG, lnB, lnMod, M, N, K);
}

extern "C" void kernel_launch(void* const* d_in, const int* in_sizes, int n_in,
                              void* d_out, int out_size, void* d_ws, size_t ws_size,
                              hipStream_t stream) {
  (void)in_sizes; (void)n_in; (void)out_size; (void)ws_size;

  const float* x    = (const float*)d_in[0];
  const float* Wp   = (const float*)d_in[1];
  const float* bp   = (const float*)d_in[2];
  const float* cls  = (const float*)d_in[3];
  const float* pos  = (const float*)d_in[4];
  const float* ln1g = (const float*)d_in[5];
  const float* ln1b = (const float*)d_in[6];
  const float* Wq   = (const float*)d_in[7];
  const float* bq   = (const float*)d_in[8];
  const float* Wk   = (const float*)d_in[9];
  const float* bk   = (const float*)d_in[10];
  const float* Wv   = (const float*)d_in[11];
  const float* bv   = (const float*)d_in[12];
  const float* Wo   = (const float*)d_in[13];
  const float* bo   = (const float*)d_in[14];
  const float* ln2g = (const float*)d_in[15];
  const float* ln2b = (const float*)d_in[16];
  const float* W1   = (const float*)d_in[17];
  const float* b1   = (const float*)d_in[18];
  const float* W2   = (const float*)d_in[19];
  const float* b2   = (const float*)d_in[20];
  const float* lnfg = (const float*)d_in[21];
  const float* lnfb = (const float*)d_in[22];
  const float* Wc   = (const float*)d_in[23];
  const float* bc   = (const float*)d_in[24];
  float* out = (float*)d_out;

  const long long S = 197, D = 768, BS = 16 * 197, SCLD = 224;

  float* ws = (float*)d_ws;
  long long o = 0;
  float* Ape = ws + o; o += 3136LL * 768;    // im2col matrix
  float* pe  = ws + o; o += 3136LL * 768;    // patch-embed output
  float* h   = ws + o; o += BS * D;          // residual stream [b*s, 768]
  float* mu  = ws + o; o += BS;
  float* rs  = ws + o; o += BS;
  float* qb  = ws + o; o += 12LL * BS * D;   // head-major [h][b*s][dk]
  float* kb  = ws + o; o += 12LL * BS * D;
  float* vb  = ws + o; o += 12LL * BS * D;
  float* sc  = ws + o; o += 192LL * S * SCLD; // scores per (b,h), ld=224
  float* pr  = ws + o; o += 192LL * S * SCLD; // probs
  float* cat = ws + o; o += BS * 9216LL;      // head-concat [b*s, H*DV]
  float* hid = ws + o; o += BS * 3072LL;      // MLP hidden

  const float scale = 0.036084391824351614f;  // 1/sqrt(768)
  const long long HS = BS * D;                // per-head q/k/v plane (3152*768)

  // patch embed: im2col + GEMM (W_patch is [N=768][K=768] row-major -> TRB)
  vit_im2col<<<(unsigned)((3136LL * 768 + 255) / 256), 256, 0, stream>>>(x, Ape);
  launch_gemm<true, false, false, false>(stream,
      Ape, 768, 1, 0, 1, 0,
      Wp, 768, 1, 0, 1, 0,
      pe, 768, 1, 0, 1, 0,
      bp, 1, 0, nullptr, nullptr, nullptr, nullptr, nullptr, 1,
      3136, 768, 768, 1);
  vit_assemble<<<(unsigned)((BS * D + 255) / 256), 256, 0, stream>>>(pe, cls, pos, h);

  for (int l = 0; l < 4; ++l) {
    const float* Wq_l = Wq + (long long)l * 12 * 768 * 768;
    const float* Wk_l = Wk + (long long)l * 12 * 768 * 768;
    const float* Wv_l = Wv + (long long)l * 12 * 768 * 768;
    const float* bq_l = bq + (long long)l * 12 * 768;
    const float* bk_l = bk + (long long)l * 12 * 768;
    const float* bv_l = bv + (long long)l * 12 * 768;
    const float* Wo_l = Wo + (long long)l * 9216 * 768;
    const float* bo_l = bo + (long long)l * 768;
    const float* g1_l = ln1g + (long long)l * 12 * 768;
    const float* e1_l = ln1b + (long long)l * 12 * 768;
    const float* g2_l = ln2g + (long long)l * 768;
    const float* e2_l = ln2b + (long long)l * 768;
    const float* W1_l = W1 + (long long)l * 768 * 3072;
    const float* b1_l = b1 + (long long)l * 3072;
    const float* W2_l = W2 + (long long)l * 3072 * 768;
    const float* b2_l = b2 + (long long)l * 768;

    // LN1 stats (shared across heads); QKV as 12 head-batched 3152x768x768
    // GEMMs with LN fused into the A load (z = head).
    vit_ln_stats<<<(unsigned)BS, 256, 0, stream>>>(h, 1, mu, rs);
    launch_gemm<false, true, false, false>(stream,
        h, 768, 1, 0, 1, 0,
        Wq_l, 768, 1, 0, 12, 768LL * 768,
        qb, 768, 1, HS, 1, 0,
        bq_l, 12, 768, nullptr, mu, rs, g1_l, e1_l, 12,
        3152, 768, 768, 12);
    launch_gemm<false, true, false, false>(stream,
        h, 768, 1, 0, 1, 0,
        Wk_l, 768, 1, 0, 12, 768LL * 768,
        kb, 768, 1, HS, 1, 0,
        bk_l, 12, 768, nullptr, mu, rs, g1_l, e1_l, 12,
        3152, 768, 768, 12);
    launch_gemm<false, true, false, false>(stream,
        h, 768, 1, 0, 1, 0,
        Wv_l, 768, 1, 0, 12, 768LL * 768,
        vb, 768, 1, HS, 1, 0,
        bv_l, 12, 768, nullptr, mu, rs, g1_l, e1_l, 12,
        3152, 768, 768, 12);

    // scores[b,h] = q[b,h] @ k[b,h]^T   (z = b*12 + h; q/k are head-major)
    launch_gemm<true, false, false, false>(stream,
        qb, 768, 12, S * D, 12, HS,
        kb, 768, 12, S * D, 12, HS,
        sc, SCLD, 1, S * SCLD, 1, 0,
        nullptr, 1, 0, nullptr, nullptr, nullptr, nullptr, nullptr, 1,
        197, 197, 768, 192);

    // softmax over the query axis, with 1/sqrt(dk) scaling
    vit_softmax_q<<<dim3(1, 192), 256, 0, stream>>>(sc, pr, 197, 224, scale);

    // av[b,h] = probs[b,h] @ v[b,h]; write directly into head-concat layout
    launch_gemm<false, false, false, false>(stream,
        pr, SCLD, 1, S * SCLD, 1, 0,
        vb, 768, 12, S * D, 12, HS,
        cat, 9216, 12, S * 9216, 12, 768,
        nullptr, 1, 0, nullptr, nullptr, nullptr, nullptr, nullptr, 1,
        197, 768, 197, 192);

    // h = h + cat @ Wo + bo
    launch_gemm<false, false, false, true>(stream,
        cat, 9216, 1, 0, 1, 0,
        Wo_l, 768, 1, 0, 1, 0,
        h, 768, 1, 0, 1, 0,
        bo_l, 1, 0, h, nullptr, nullptr, nullptr, nullptr, 1,
        3152, 768, 9216, 1);

    // LN2 + MLP (GELU fused in first GEMM, residual fused in second)
    vit_ln_stats<<<(unsigned)BS, 256, 0, stream>>>(h, 1, mu, rs);
    launch_gemm<false, true, true, false>(stream,
        h, 768, 1, 0, 1, 0,
        W1_l, 3072, 1, 0, 1, 0,
        hid, 3072, 1, 0, 1, 0,
        b1_l, 1, 0, nullptr, mu, rs, g2_l, e2_l, 1,
        3152, 3072, 768, 1);
    launch_gemm<false, false, false, true>(stream,
        hid, 3072, 1, 0, 1, 0,
        W2_l, 768, 1, 0, 1, 0,
        h, 768, 1, 0, 1, 0,
        b2_l, 1, 0, h, nullptr, nullptr, nullptr, nullptr, 1,
        3152, 768, 3072, 1);
  }

  // final LN on CLS rows + classifier
  vit_ln_stats<<<16, 256, 0, stream>>>(h, 197, mu, rs);
  vit_classifier<<<1, 64, 0, stream>>>(h, mu, rs, lnfg, lnfb, Wc, bc, out);
}
